// MultiHeadSelfAttention_37692632990145
// MI455X (gfx1250) — matrix operand order repaired and checked by reading, not compiled, not run on hardware
//
#include <hip/hip_runtime.h>
#include <hip/hip_bf16.h>

// ---------------------------------------------------------------------------
// MultiHeadSelfAttention  B=4, N=2048, D=768, H=12, HD=64  — CDNA5 WMMA f16
//   - QKV / out-proj GEMMs: 4-wave workgroups, 64x128 tiles, k-step 64,
//     double-buffered GLOBAL_LOAD_ASYNC_TO_LDS staging (ASYNCcnt),
//     software-pipelined ds_load_b128 -> WMMA bursts
//   - attention: flash-style online softmax, one wave per 16-query tile
// ---------------------------------------------------------------------------

typedef __attribute__((ext_vector_type(16))) _Float16 v16h;
typedef __attribute__((ext_vector_type(8)))  _Float16 v8h;
typedef __attribute__((ext_vector_type(8)))  float    v8f;
typedef __attribute__((ext_vector_type(4)))  int      v4i;

typedef __attribute__((address_space(1))) v4i as1_v4i;
typedef __attribute__((address_space(3))) v4i as3_v4i;

#define DEV __device__ __forceinline__

constexpr int Bc  = 4;
constexpr int Ncq = 2048;
constexpr int Dc  = 768;
constexpr int Hc  = 12;
constexpr int HDc = 64;
constexpr float SCALE_Q = 0.125f;   // 64^-0.5

// GEMM LDS tile: 64 k-halves + 8 pad = 144B rows (16B-aligned, conflict-free)
constexpr int GEMM_STRIDE = 72;
// attention P tile: 32 k-halves + 8 pad = 80B rows
constexpr int ATT_STRIDE = 40;

// ---- CDNA5 async copy / wait -----------------------------------------------

DEV void async_copy16(const _Float16* gsrc, unsigned lds_off) {
#if __has_builtin(__builtin_amdgcn_global_load_async_to_lds_b128)
  __builtin_amdgcn_global_load_async_to_lds_b128(
      (as1_v4i*)(unsigned long long)(size_t)gsrc,
      (as3_v4i*)lds_off, 0, 0);
#else
  asm volatile("global_load_async_to_lds_b128 %0, %1, off"
               :: "v"(lds_off), "v"((unsigned long long)(size_t)gsrc)
               : "memory");
#endif
}

#if __has_builtin(__builtin_amdgcn_s_wait_asynccnt)
#define WAIT_ASYNC(n) __builtin_amdgcn_s_wait_asynccnt(n)
#else
#define WAIT_ASYNC(n) asm volatile("s_wait_asynccnt %0" :: "i"(n) : "memory")
#endif

// low 32 bits of a generic pointer to __shared__ == LDS byte offset
DEV unsigned lds_off_of(const void* p) { return (unsigned)(size_t)p; }

// ---- WMMA helpers ----------------------------------------------------------

DEV v8f wmma_f16(v16h a, v16h b, v8f c) {
  return __builtin_amdgcn_wmma_f32_16x16x32_f16(
      false, a, false, b, (short)0, c, false, false);
}

// A fragment (16x32 f16) from row-major source, row = lane&15.
// lanes 0-15: K = k0+[0..15]; lanes 16-31: K = k0+16+[0..15].
DEV v16h load_a_frag(const _Float16* row_base, int ld, int k0, int lane) {
  const int cl = lane & 15, gh = lane >> 4;
  const _Float16* rp = row_base + (size_t)cl * ld + k0;
  v8h lo = *(const v8h*)(rp + 16 * gh);
  v8h hi = *(const v8h*)(rp + 16 * gh + 8);
  v16h a;
#pragma unroll
  for (int i = 0; i < 8; ++i) { a[i] = lo[i]; a[i + 8] = hi[i]; }
  return a;
}

// B fragment (32x16 f16) from source where column n has contiguous K.
// col = lane&15, K = 16*(lane>>4) + e.
DEV v16h load_b_frag(const _Float16* colmaj_base, int ldk, int n0, int k0, int lane) {
  const int cl = lane & 15, gh = lane >> 4;
  return *(const v16h*)(colmaj_base + (size_t)(n0 + cl) * ldk + k0 + 16 * gh);
}

// same fragments out of an LDS tile with `stride` row pitch (halves)
DEV v16h lds_a_frag(const _Float16* buf, int stride, int row0, int k0, int lane) {
  const int cl = lane & 15, gh = lane >> 4;
  const _Float16* rp = buf + (row0 + cl) * stride + k0;
  v8h lo = *(const v8h*)(rp + 16 * gh);
  v8h hi = *(const v8h*)(rp + 16 * gh + 8);
  v16h a;
#pragma unroll
  for (int i = 0; i < 8; ++i) { a[i] = lo[i]; a[i + 8] = hi[i]; }
  return a;
}
DEV v16h lds_b_frag(const _Float16* buf, int stride, int col0, int k0, int lane) {
  const int cl = lane & 15, gh = lane >> 4;
  return *(const v16h*)(buf + (col0 + cl) * stride + k0 + 16 * gh);
}

// 16-lane-segment reductions
DEV float rmax16(float x) {
  x = fmaxf(x, __shfl_xor(x, 1, 16));
  x = fmaxf(x, __shfl_xor(x, 2, 16));
  x = fmaxf(x, __shfl_xor(x, 4, 16));
  x = fmaxf(x, __shfl_xor(x, 8, 16));
  return x;
}
DEV float rsum16(float x) {
  x += __shfl_xor(x, 1, 16);
  x += __shfl_xor(x, 2, 16);
  x += __shfl_xor(x, 4, 16);
  x += __shfl_xor(x, 8, 16);
  return x;
}

// ---- small prep kernels ----------------------------------------------------

__global__ void cast_f32_to_f16_kernel(const float* __restrict__ src,
                                       _Float16* __restrict__ dst, int n) {
  int i = blockIdx.x * blockDim.x + threadIdx.x;
  if (i < n) dst[i] = (_Float16)src[i];
}

// dst[n*K + k] = (f16) src[k*Ncols + n]
__global__ void transpose_to_f16_kernel(const float* __restrict__ src,
                                        _Float16* __restrict__ dst,
                                        int K, int Ncols) {
  int i = blockIdx.x * blockDim.x + threadIdx.x;
  if (i >= K * Ncols) return;
  int n = i / K;
  int k = i - n * K;
  dst[i] = (_Float16)src[(size_t)k * Ncols + n];
}

// ---- WMMA GEMM with double-buffered async-to-LDS staging -------------------
// Workgroup = 128 threads (4 waves), tile 64(M) x 128(N), staged k-step 64.
// A row-major f16; Bt column-major f16 (Bt[n*K+k]).
// mode 0: route QKV -> Qh/Kh (B,H,N,HD), Vt (B,H,HD,N).  mode 1: f32 Out.

// per k-step: A 64x64 halves (512 16B-chunks) + B 128x64 (1024 chunks)
// 1536 chunks over 128 threads -> 12 async b128 per thread.
DEV void stage_tiles(const _Float16* __restrict__ A,
                     const _Float16* __restrict__ Bt, int K,
                     int m0, int n0, int k0, int t,
                     unsigned abase, unsigned bbase) {
#pragma unroll
  for (int j = 0; j < 12; ++j) {
    const int chunk = j * 128 + t;
    if (chunk < 512) {
      const int row = chunk >> 3, sub = chunk & 7;
      async_copy16(A + (size_t)(m0 + row) * K + k0 + sub * 8,
                   abase + (unsigned)(row * GEMM_STRIDE + sub * 8) * 2u);
    } else {
      const int c = chunk - 512;
      const int col = c >> 3, sub = c & 7;
      async_copy16(Bt + (size_t)(n0 + col) * K + k0 + sub * 8,
                   bbase + (unsigned)(col * GEMM_STRIDE + sub * 8) * 2u);
    }
  }
}

__global__ void __launch_bounds__(128)
wmma_gemm_kernel(const _Float16* __restrict__ A, const _Float16* __restrict__ Bt,
                 const float* __restrict__ bias, int M, int K, int NC, int mode,
                 _Float16* __restrict__ Qh, _Float16* __restrict__ Kh,
                 _Float16* __restrict__ Vt, float* __restrict__ Out) {
  const int t    = threadIdx.x;
  const int wave = t >> 5;
  const int lane = t & 31;
  const int cl = lane & 15, gh = lane >> 4;
  const int m0 = blockIdx.x * 64;
  const int n0 = blockIdx.y * 128;

  __shared__ __align__(16) _Float16 Abuf[2][64  * GEMM_STRIDE];
  __shared__ __align__(16) _Float16 Bbuf[2][128 * GEMM_STRIDE];
  const unsigned aoff[2] = {lds_off_of(&Abuf[0][0]), lds_off_of(&Abuf[1][0])};
  const unsigned boff[2] = {lds_off_of(&Bbuf[0][0]), lds_off_of(&Bbuf[1][0])};

  v8f zero = {};
  v8f acc[8];
#pragma unroll
  for (int f = 0; f < 8; ++f) acc[f] = zero;

  const int steps = K / 64;
  stage_tiles(A, Bt, K, m0, n0, 0, t, aoff[0], boff[0]);

  for (int i = 0; i < steps; ++i) {
    const int cur = i & 1;
    const bool more = (i + 1) < steps;
    if (more)
      stage_tiles(A, Bt, K, m0, n0, (i + 1) * 64, t, aoff[cur ^ 1], boff[cur ^ 1]);
    if (more) { WAIT_ASYNC(12); } else { WAIT_ASYNC(0); }
    __syncthreads();                       // tile i fully in LDS

#pragma unroll
    for (int kk = 0; kk < 64; kk += 32) {
      v16h a = lds_a_frag(&Abuf[cur][0], GEMM_STRIDE, wave * 16, kk, lane);
      // 2-deep software pipeline: issue loads for f+2 before the WMMA for f,
      // so each WMMA waits on loads issued two fragments earlier.
      v16h bcur = lds_b_frag(&Bbuf[cur][0], GEMM_STRIDE, 0, kk, lane);
      v16h bnxt = lds_b_frag(&Bbuf[cur][0], GEMM_STRIDE, 16, kk, lane);
#pragma unroll
      for (int f = 0; f < 8; ++f) {
        v16h bld = bcur;
        if (f + 2 < 8)
          bld = lds_b_frag(&Bbuf[cur][0], GEMM_STRIDE, (f + 2) * 16, kk, lane);
        acc[f] = wmma_f16(a, bcur, acc[f]);
        bcur = bnxt;
        bnxt = bld;
      }
    }
    __syncthreads();                       // reads of tile i done before overwrite
  }

  // ---- epilogue -----------------------------------------------------------
#pragma unroll
  for (int f = 0; f < 8; ++f) {
    const int col = n0 + f * 16 + cl;
    const float bv = bias[col];
#pragma unroll
    for (int rr = 0; rr < 8; ++rr) {
      const int row = m0 + wave * 16 + 8 * gh + rr;
      const float v = acc[f][rr] + bv;
      if (mode == 0) {
        const int which = col / Dc;            // 0=Q 1=K 2=V
        const int cc = col - which * Dc;
        const int h = cc >> 6, hd = cc & 63;
        const int b = row >> 11, n = row & (Ncq - 1);
        const size_t idx = ((size_t)(b * Hc + h) * Ncq + n) * HDc + hd;
        if (which == 0)       Qh[idx] = (_Float16)(v * SCALE_Q);
        else if (which == 1)  Kh[idx] = (_Float16)v;
        else Vt[((size_t)(b * Hc + h) * HDc + hd) * Ncq + n] = (_Float16)v;
      } else {
        Out[(size_t)row * NC + col] = v;
      }
    }
  }
}

// ---- flash attention: one wave = 16 query rows of one (b,h) ----------------
__global__ void __launch_bounds__(32)
attention_kernel(const _Float16* __restrict__ Qh, const _Float16* __restrict__ Kh,
                 const _Float16* __restrict__ Vt, _Float16* __restrict__ O) {
  const int lane = threadIdx.x;
  const int cl = lane & 15, gh = lane >> 4;
  const int qt = blockIdx.x;                 // 16-query tile
  const int bh = blockIdx.y;
  const int b = bh / Hc, h = bh - b * Hc;

  const _Float16* Qbase = Qh + ((size_t)bh * Ncq + qt * 16) * HDc;
  const _Float16* Kbase = Kh + (size_t)bh * Ncq * HDc;   // (key, hd)
  const _Float16* Vbase = Vt + (size_t)bh * HDc * Ncq;   // (hd, key)

  const v16h qa0 = load_a_frag(Qbase, HDc, 0, lane);
  const v16h qa1 = load_a_frag(Qbase, HDc, 32, lane);

  v8f zero = {};
  v8f acc[4] = {zero, zero, zero, zero};
  float mrow[8], lrow[8];
#pragma unroll
  for (int rr = 0; rr < 8; ++rr) { mrow[rr] = -3.0e38f; lrow[rr] = 0.0f; }

  __shared__ __align__(16) _Float16 ptile[16 * ATT_STRIDE];

  for (int j0 = 0; j0 < Ncq; j0 += 32) {
    v8f s0 = zero, s1 = zero;
    {
      v16h kb00 = load_b_frag(Kbase, HDc, j0,      0,  lane);
      v16h kb01 = load_b_frag(Kbase, HDc, j0,      32, lane);
      v16h kb10 = load_b_frag(Kbase, HDc, j0 + 16, 0,  lane);
      v16h kb11 = load_b_frag(Kbase, HDc, j0 + 16, 32, lane);
      s0 = wmma_f16(qa0, kb00, s0);
      s0 = wmma_f16(qa1, kb01, s0);
      s1 = wmma_f16(qa0, kb10, s1);
      s1 = wmma_f16(qa1, kb11, s1);
    }

#pragma unroll
    for (int rr = 0; rr < 8; ++rr) {
      float tm = rmax16(fmaxf(s0[rr], s1[rr]));
      float mnew = fmaxf(mrow[rr], tm);
      float cf = __expf(mrow[rr] - mnew);
      float p0 = __expf(s0[rr] - mnew);
      float p1 = __expf(s1[rr] - mnew);
      float rs = rsum16(p0 + p1);
      lrow[rr] = lrow[rr] * cf + rs;
      mrow[rr] = mnew;
#pragma unroll
      for (int f = 0; f < 4; ++f) acc[f][rr] *= cf;
      const int prow = 8 * gh + rr;
      ptile[prow * ATT_STRIDE + cl]      = (_Float16)p0;
      ptile[prow * ATT_STRIDE + 16 + cl] = (_Float16)p1;
    }
    __syncthreads();

    v16h pa;
    {
      const _Float16* pr = &ptile[cl * ATT_STRIDE];
      v8h lo = *(const v8h*)(pr + 16 * gh);
      v8h hi = *(const v8h*)(pr + 16 * gh + 8);
#pragma unroll
      for (int i = 0; i < 8; ++i) { pa[i] = lo[i]; pa[i + 8] = hi[i]; }
    }
    __syncthreads();

#pragma unroll
    for (int f = 0; f < 4; ++f) {
      v16h vb = *(const v16h*)(Vbase + (size_t)(f * 16 + cl) * Ncq + j0 + 16 * gh);
      acc[f] = wmma_f16(pa, vb, acc[f]);
    }
  }

#pragma unroll
  for (int rr = 0; rr < 8; ++rr) {
    const float inv = 1.0f / lrow[rr];
    const int row = qt * 16 + 8 * gh + rr;
#pragma unroll
    for (int f = 0; f < 4; ++f) {
      O[(size_t)(b * Ncq + row) * Dc + h * HDc + f * 16 + cl] =
          (_Float16)(acc[f][rr] * inv);
    }
  }
}

// ---------------------------------------------------------------------------

extern "C" void kernel_launch(void* const* d_in, const int* in_sizes, int n_in,
                              void* d_out, int out_size, void* d_ws, size_t ws_size,
                              hipStream_t stream) {
  (void)in_sizes; (void)n_in; (void)out_size; (void)ws_size;

  const float* x      = (const float*)d_in[0];   // (B,N,D)
  const float* w_qkv  = (const float*)d_in[1];   // (D,3D)
  const float* b_qkv  = (const float*)d_in[2];   // (3D)
  const float* w_proj = (const float*)d_in[3];   // (D,D)
  const float* b_proj = (const float*)d_in[4];   // (D)
  float* out = (float*)d_out;                    // (B,N,D) f32

  const size_t MN = (size_t)Bc * Ncq;            // 8192 rows

  char* ws = (char*)d_ws;
  size_t off = 0;
  auto alloc = [&](size_t bytes) -> char* {
    char* p = ws + off;
    off = (off + bytes + 255) & ~(size_t)255;
    return p;
  };
  _Float16* xh     = (_Float16*)alloc(MN * Dc * sizeof(_Float16));
  _Float16* wqkvT  = (_Float16*)alloc((size_t)3 * Dc * Dc * sizeof(_Float16));
  _Float16* wprojT = (_Float16*)alloc((size_t)Dc * Dc * sizeof(_Float16));
  _Float16* Qh     = (_Float16*)alloc((size_t)Bc * Hc * Ncq * HDc * sizeof(_Float16));
  _Float16* Kh     = (_Float16*)alloc((size_t)Bc * Hc * Ncq * HDc * sizeof(_Float16));
  _Float16* Vt     = (_Float16*)alloc((size_t)Bc * Hc * HDc * Ncq * sizeof(_Float16));
  _Float16* Oh     = (_Float16*)alloc(MN * Dc * sizeof(_Float16));

  {
    int n = (int)(MN * Dc);
    cast_f32_to_f16_kernel<<<(n + 255) / 256, 256, 0, stream>>>(x, xh, n);
  }
  {
    int n = Dc * 3 * Dc;
    transpose_to_f16_kernel<<<(n + 255) / 256, 256, 0, stream>>>(w_qkv, wqkvT, Dc, 3 * Dc);
    int n2 = Dc * Dc;
    transpose_to_f16_kernel<<<(n2 + 255) / 256, 256, 0, stream>>>(w_proj, wprojT, Dc, Dc);
  }

  // QKV projection GEMM: 8192 x 768 x 2304
  wmma_gemm_kernel<<<dim3((unsigned)(MN / 64), (3 * Dc) / 128), 128, 0, stream>>>(
      xh, wqkvT, b_qkv, (int)MN, Dc, 3 * Dc, /*mode=*/0, Qh, Kh, Vt, nullptr);

  // flash attention: 128 query tiles x 48 (b,h)
  attention_kernel<<<dim3(Ncq / 16, Bc * Hc), 32, 0, stream>>>(Qh, Kh, Vt, Oh);

  // output projection GEMM: 8192 x 768 x 768
  wmma_gemm_kernel<<<dim3((unsigned)(MN / 64), Dc / 128), 128, 0, stream>>>(
      Oh, wprojT, b_proj, (int)MN, Dc, Dc, /*mode=*/1,
      nullptr, nullptr, nullptr, out);
}
